// Net_10866267259242
// MI455X (gfx1250) — matrix operand order repaired and checked by reading, not compiled, not run on hardware
//
#include <hip/hip_runtime.h>

// ---------------------------------------------------------------------------
// SNN Leaky (snntorch-style) fused GEMM + LIF scan for MI455X (gfx1250).
//   cur[t] = x[t] @ W^T           (bf16 WMMA, f32 accumulate)
//   mem    = beta*mem + cur - (mem_prev>1)*1 ;  spk = (mem>1)
//
// Fusion keeps membrane state in accumulator VGPRs for all 100 timesteps.
// K is split in half across wave pairs (LDS reduction per timestep) so the
// dispatch exposes 1024 concurrent waves instead of 512.
// ---------------------------------------------------------------------------

#define T_STEPS 100
#define BATCH   128
#define NI      2048
#define NO      1024
#define KSPLIT  (NI / 2)     // 1024 per K-half
#define BETA    0.9f
#define THRESH  1.0f

typedef __attribute__((ext_vector_type(16))) __bf16 v16bf;
typedef __attribute__((ext_vector_type(8)))  __bf16 v8bf;
typedef __attribute__((ext_vector_type(4)))  __bf16 v4bf;
typedef __attribute__((ext_vector_type(8)))  float  v8f;

// ---------------------------------------------------------------------------
// f32 -> bf16 conversion (grid-stride, 4-wide). n must be a multiple of 4.
// ---------------------------------------------------------------------------
__global__ void cvt_f32_bf16(const float* __restrict__ src,
                             __bf16* __restrict__ dst, int n) {
  int i = (blockIdx.x * blockDim.x + threadIdx.x) * 4;
  const int stride = gridDim.x * blockDim.x * 4;
  for (; i < n; i += stride) {
    float4 v = *(const float4*)(src + i);
    v4bf o;
    o.x = (__bf16)v.x; o.y = (__bf16)v.y;
    o.z = (__bf16)v.z; o.w = (__bf16)v.w;
    *(v4bf*)(dst + i) = o;
  }
}

// ---------------------------------------------------------------------------
// Fused kernel.
//   block = 256 threads (8 wave32). Block tile = 16(batch) x 64(neuron).
//   wave w: ntile = w&3 (16 neurons), ktile = w>>2 (K half).
//   ktile==1 waves write partial accumulators to LDS; ktile==0 waves reduce,
//   run the LIF update, and store spk/mem. One barrier per timestep using a
//   double-buffered reduction area.
//   Grid = (NO/64, BATCH/16) = (16, 8) -> 128 blocks, 1024 waves.
// ---------------------------------------------------------------------------
__global__ __launch_bounds__(256, 1)
void snn_fused(const __bf16* __restrict__ xb,   // [T][B][NI] bf16
               const __bf16* __restrict__ wb,   // [NO][NI]   bf16
               float* __restrict__ out) {       // spk[T][B][NO] ++ mem[T][B][NO]
  // 48B lane stride: 16B-aligned (b128 ds ops), bank-conflict-spread.
  __shared__ float red[2][4][32][12];

  const int lane  = threadIdx.x & 31;
  const int wave  = threadIdx.x >> 5;
  const int ntile = wave & 3;          // which 16-neuron column
  const int ktile = wave >> 2;         // which K half
  const int llo   = lane & 15;
  const int hi    = lane >> 4;

  const int o0 = blockIdx.x * 64 + ntile * 16;  // neuron tile base
  const int b0 = blockIdx.y * 16;               // batch  tile base

  // B-matrix (W^T) fragment source: lane llo supplies column (o0+llo);
  // lanes 0-15 hold K=0..15, lanes 16-31 K=16..31 -> contiguous 32B run.
  const __bf16* wrow  = wb + (size_t)(o0 + llo) * NI + ktile * KSPLIT + hi * 16;
  // A-matrix (x) fragment source: lane llo supplies row (b0+llo).
  const __bf16* xbase = xb + (size_t)(b0 + llo) * NI + ktile * KSPLIT;

  v8f mem = {};                                 // membrane state (ktile==0)
  float* __restrict__ spk_out = out;
  float* __restrict__ mem_out = out + (size_t)T_STEPS * BATCH * NO;

  for (int t = 0; t < T_STEPS; ++t) {
    const __bf16* xrow = xbase + (size_t)t * BATCH * NI;

    // Warm next timestep's x slice in L2 (global_prefetch_b8).
    if (t + 1 < T_STEPS)
      __builtin_prefetch(xrow + (size_t)BATCH * NI, 0, 1);

    // Two independent accumulator chains: insensitive to XDL accumulate-
    // forwarding depth (D->C needs no NOPs per ISA 7.12.1, but dual chains
    // guarantee full issue rate).
    v8f a0 = {}, a1 = {};
#pragma unroll 4
    for (int k = 0; k < KSPLIT; k += 64) {
      // A 16x32 layout (matches B): lane(lo) K = 0..15,
      //                             lane(hi) K = 16..31
      union { v16bf v; v8bf h[2]; } A0, B0, A1, B1;
      A0.h[0] = *(const v8bf*)(xrow + k + hi * 16);
      A0.h[1] = *(const v8bf*)(xrow + k + 8 + hi * 16);
      B0.h[0] = *(const v8bf*)(wrow + k);
      B0.h[1] = *(const v8bf*)(wrow + k + 8);
      A1.h[0] = *(const v8bf*)(xrow + k + 32 + hi * 16);
      A1.h[1] = *(const v8bf*)(xrow + k + 40 + hi * 16);
      B1.h[0] = *(const v8bf*)(wrow + k + 32);
      B1.h[1] = *(const v8bf*)(wrow + k + 40);
      a0 = __builtin_amdgcn_wmma_f32_16x16x32_bf16(
          false, A0.v, false, B0.v, (short)0, a0, false, false);
      a1 = __builtin_amdgcn_wmma_f32_16x16x32_bf16(
          false, A1.v, false, B1.v, (short)0, a1, false, false);
    }
    v8f acc = a0 + a1;

    // Cross-wave K reduction: upper-K waves publish, lower-K waves consume.
    float* slot = &red[t & 1][ntile][lane][0];
    if (ktile == 1) {
#pragma unroll
      for (int j = 0; j < 8; ++j) slot[j] = acc[j];
    }
    __syncthreads();

    if (ktile == 0) {
#pragma unroll
      for (int j = 0; j < 8; ++j) acc[j] += slot[j];

      // LIF update + store. Accumulator layout: VGPR j holds
      // (b = b0 + j + 8*hi, o = o0 + (lane&15)).
      const int bbase = b0 + 8 * hi;
      const int o     = o0 + llo;
#pragma unroll
      for (int j = 0; j < 8; ++j) {
        float m     = mem[j];
        float reset = (m > THRESH) ? THRESH : 0.0f;   // heaviside(mem_prev-1)
        m           = BETA * m + acc[j] - reset;
        float spk   = (m > THRESH) ? 1.0f : 0.0f;     // heaviside(mem_new-1)
        mem[j]      = m;
        const size_t idx = ((size_t)t * BATCH + (bbase + j)) * NO + o;
        spk_out[idx] = spk;
        mem_out[idx] = m;
      }
    }
    // No second barrier: double-buffered `red` makes write(t+2) of a buffer
    // ordered after the barrier at t+1, which follows the reads at t.
  }
}

// ---------------------------------------------------------------------------
// Workspace layout:
//   [0 .. 52,428,800)              xb : [T][B][NI] bf16
//   [52,428,800 .. 56,623,104)     wb : [NO][NI]   bf16
// ---------------------------------------------------------------------------
extern "C" void kernel_launch(void* const* d_in, const int* in_sizes, int n_in,
                              void* d_out, int out_size, void* d_ws, size_t ws_size,
                              hipStream_t stream) {
  (void)in_sizes; (void)n_in; (void)out_size; (void)ws_size;
  const float* x = (const float*)d_in[0];   // [T,B,NI] f32
  const float* W = (const float*)d_in[1];   // [NO,NI]  f32

  const int nx = T_STEPS * BATCH * NI;      // 26,214,400
  const int nw = NO * NI;                   //  2,097,152

  __bf16* xb = (__bf16*)d_ws;
  __bf16* wb = (__bf16*)((char*)d_ws + (size_t)nx * sizeof(__bf16));

  cvt_f32_bf16<<<2048, 256, 0, stream>>>(x, xb, nx);
  cvt_f32_bf16<<<512,  256, 0, stream>>>(W, wb, nw);

  dim3 grid(NO / 64, BATCH / 16);           // (16, 8) = 128 workgroups
  snn_fused<<<grid, 256, 0, stream>>>(xb, wb, (float*)d_out);
}
